// LLaMAAttention_76733885710477
// MI455X (gfx1250) — compile-verified, matrix operand order checked by reading, not run on hardware
//
#include <hip/hip_runtime.h>

// ---------------- problem constants ----------------
constexpr int S_LEN = 2048;
constexpr int HID   = 4096;
constexpr int NH    = 32;
constexpr int NKV   = 8;
constexpr int HD    = 128;
constexpr int GQA   = NH / NKV;           // 4
constexpr float SCALING = 0.08838834764831845f;   // 128^-0.5

// ---------------- types ----------------
typedef __bf16 bf16;
typedef __attribute__((ext_vector_type(16))) __bf16 v16bf;
typedef __attribute__((ext_vector_type(4)))  __bf16 v4bf;
typedef __attribute__((ext_vector_type(8)))  float  v8f;
typedef __attribute__((ext_vector_type(4)))  unsigned int v4u;
typedef __attribute__((ext_vector_type(4)))  int v4i;
typedef __attribute__((ext_vector_type(8)))  int v8i;

__device__ __forceinline__ bf16 f2bf(float x) {
    unsigned u = __builtin_bit_cast(unsigned, x);
    unsigned r = (u + 0x7FFFu + ((u >> 16) & 1u)) >> 16;   // RNE
    return __builtin_bit_cast(bf16, (unsigned short)r);
}
__device__ __forceinline__ float bf2f(bf16 b) {
    unsigned u = ((unsigned)__builtin_bit_cast(unsigned short, b)) << 16;
    return __builtin_bit_cast(float, u);
}
__device__ __forceinline__ void storeC(float* p, float v) { *p = v; }
__device__ __forceinline__ void storeC(bf16*  p, float v) { *p = f2bf(v); }

__device__ __forceinline__ v8f wmma_bf16(v16bf a, v16bf b, v8f c) {
    return __builtin_amdgcn_wmma_f32_16x16x32_bf16(false, a, false, b, (short)0, c, false, false);
}

// ---------------- CDNA5 async global->LDS staging ----------------
__device__ __forceinline__ void async_ld_b128(void* lds_dst, const void* gsrc) {
    unsigned lo = (unsigned)(size_t)lds_dst;                // low 32 bits = LDS offset
    unsigned long long ga = (unsigned long long)gsrc;
    asm volatile("global_load_async_to_lds_b128 %0, %1, off"
                 :: "v"(lo), "v"(ga) : "memory");
}
__device__ __forceinline__ void wait_async0() {
    asm volatile("s_wait_asynccnt 0" ::: "memory");
}
__device__ __forceinline__ void wait_async6() {
    asm volatile("s_wait_asynccnt 6" ::: "memory");         // newest batch (6/thread) may stay in flight
}

// ---------------- TDM (Tensor Data Mover) K/V tile load --------------------
// 2D tile: 128 contiguous bf16 (x) by 32 rows (y), row stride in elements.
// LDS padding: 4 DWORDs (8 bf16) every 64 DWORDs (128 bf16) -> stride HD+8.
#if __has_builtin(__builtin_amdgcn_tensor_load_to_lds) && __has_builtin(__builtin_amdgcn_s_wait_tensorcnt)
#define HAVE_TDM 1
__device__ __forceinline__ void tdm_load_tile_128x32(void* lds_dst, const bf16* gptr,
                                                     int row_stride_elems) {
    unsigned long long ga = (unsigned long long)(const void*)gptr;
    v4u g0;
    g0[0] = 1u;                                          // count=1, user desc
    g0[1] = (unsigned)(size_t)lds_dst;                   // lds_addr
    g0[2] = (unsigned)(ga & 0xFFFFFFFFu);                // global_addr[31:0]
    g0[3] = (unsigned)((ga >> 32) & 0x1FFFFFFu) | (2u << 30);  // addr[56:32], type=2
    v8i g1;
    g1[0] = (int)((1u << 16)            // data_size = 2B
                | (1u << 20)            // pad_enable
                | (5u << 22)            // pad_interval: 64 DWORDs
                | (3u << 25));          // pad_amount: 4 DWORDs
    g1[1] = (int)(128u << 16);          // tensor_dim0 = 128
    g1[2] = (int)(32u  << 16);          // tensor_dim1 = 32
    g1[3] = (int)(128u << 16);          // tile_dim0 = 128
    g1[4] = 32;                         // tile_dim1 = 32, tile_dim2 = 0
    g1[5] = row_stride_elems;           // tensor_dim0_stride[31:0]
    g1[6] = 0;
    g1[7] = 0;
    v4i z = {};
#if __clang_major__ >= 23
    v8i z8 = {};
    __builtin_amdgcn_tensor_load_to_lds(g0, g1, z, z, z8, 0);
#else
    __builtin_amdgcn_tensor_load_to_lds(g0, g1, z, z, 0);
#endif
}
#endif

// ---------------- WMMA fragment loaders (16-bit A/B ISA layout) -----------
__device__ __forceinline__ v16bf frag_ld(const bf16* base, int ld) {
    int lane = threadIdx.x & 31;
    int r  = lane & 15;
    int kb = (lane & 16) ? 8 : 0;
    v16bf f;
#pragma unroll
    for (int e = 0; e < 8; ++e) {
        f[e]     = base[r * ld + kb + e];
        f[e + 8] = base[r * ld + kb + 16 + e];
    }
    return f;
}
__device__ __forceinline__ v16bf frag_ld_kn(const bf16* base, int ld) {
    int lane = threadIdx.x & 31;
    int n  = lane & 15;
    int kb = (lane & 16) ? 8 : 0;
    v16bf f;
#pragma unroll
    for (int e = 0; e < 8; ++e) {
        f[e]     = base[(kb + e)      * ld + n];
        f[e + 8] = base[(kb + 16 + e) * ld + n];
    }
    return f;
}

// ---------------- 1) fp32 -> bf16 convert (vectorized) --------------------
__global__ void cvt_f32_bf16(const float* __restrict__ in, bf16* __restrict__ out, int n4) {
    int i = blockIdx.x * blockDim.x + threadIdx.x;
    int stride = gridDim.x * blockDim.x;
    const float4* in4 = (const float4*)in;
    v4bf* out4 = (v4bf*)out;
    for (; i < n4; i += stride) {
        float4 f = in4[i];
        v4bf o; o[0] = f2bf(f.x); o[1] = f2bf(f.y); o[2] = f2bf(f.z); o[3] = f2bf(f.w);
        out4[i] = o;
    }
}

// ---------------- 2) bf16 WMMA GEMM: C[M,N] = sum_k A[m,k]*B[n,k] ----------
// block: 256 threads = 8 waves; tile 64(M) x 128(N); K-step 64.
// Double-buffered async->LDS staging overlapped with 8 WMMAs/wave/step.
template <typename OutT>
__global__ __launch_bounds__(256)
void gemm_bf16_wmma(const bf16* __restrict__ A, const bf16* __restrict__ B,
                    OutT* __restrict__ C, int M, int N, int K) {
    __shared__ __align__(16) bf16 As[2][64][72];
    __shared__ __align__(16) bf16 Bs[2][128][72];

    const int tid  = threadIdx.x;
    const int lane = tid & 31;
    const int wave = tid >> 5;            // 0..7
    const int wm = wave >> 2;             // 0..1 : 32-row strip
    const int wn = wave & 3;              // 0..3 : 32-col strip
    const int m0 = blockIdx.y * 64;
    const int n0 = blockIdx.x * 128;

    // stage one 64-wide K-slab into buffer `buf` (6 async b128 per thread)
    auto stage = [&](int buf, int k0) {
#pragma unroll
        for (int t = 0; t < 2; ++t) {                     // A: 64x64 = 512 chunks
            int idx = tid + t * 256;
            int r = idx >> 3, c = (idx & 7) * 8;
            async_ld_b128(&As[buf][r][c], A + (size_t)(m0 + r) * K + k0 + c);
        }
#pragma unroll
        for (int t = 0; t < 4; ++t) {                     // B: 128x64 = 1024 chunks
            int idx = tid + t * 256;
            int r = idx >> 3, c = (idx & 7) * 8;
            async_ld_b128(&Bs[buf][r][c], B + (size_t)(n0 + r) * K + k0 + c);
        }
    };

    v8f acc00 = {}, acc01 = {}, acc10 = {}, acc11 = {};

    stage(0, 0);
    int it = 0;
    for (int k0 = 0; k0 < K; k0 += 64, ++it) {
        const int cur = it & 1;
        const bool more = (k0 + 64 < K);
        if (more) stage(cur ^ 1, k0 + 64);   // prefetch next slab while computing
        if (more) wait_async6(); else wait_async0();
        __syncthreads();

#pragma unroll
        for (int kk = 0; kk < 64; kk += 32) {
            v16bf a0 = frag_ld(&As[cur][wm * 32][kk],      72);
            v16bf a1 = frag_ld(&As[cur][wm * 32 + 16][kk], 72);
            v16bf b0 = frag_ld(&Bs[cur][wn * 32][kk],      72);
            v16bf b1 = frag_ld(&Bs[cur][wn * 32 + 16][kk], 72);
            acc00 = wmma_bf16(a0, b0, acc00);
            acc01 = wmma_bf16(a0, b1, acc01);
            acc10 = wmma_bf16(a1, b0, acc10);
            acc11 = wmma_bf16(a1, b1, acc11);
        }
        __syncthreads();   // reads done before this buffer is refilled next iter
    }

    const int nn = n0 + wn * 32 + (lane & 15);
    const int mb = m0 + wm * 32 + ((lane >> 4) << 3);
#pragma unroll
    for (int v = 0; v < 8; ++v) {
        storeC(&C[(size_t)(mb + v) * N + nn],           acc00[v]);
        storeC(&C[(size_t)(mb + v) * N + nn + 16],      acc01[v]);
        storeC(&C[(size_t)(mb + 16 + v) * N + nn],      acc10[v]);
        storeC(&C[(size_t)(mb + 16 + v) * N + nn + 16], acc11[v]);
    }
}

// ---------------- 3) llama3-scaled RoPE (in place on bf16) ----------------
__global__ void rope_kernel(bf16* __restrict__ x, int nheads, const int* __restrict__ pos_ids) {
    int total  = S_LEN * nheads * (HD / 2);
    int stride = gridDim.x * blockDim.x;
    for (int i = blockIdx.x * blockDim.x + threadIdx.x; i < total; i += stride) {
        int d = i & 63;
        int h = (i >> 6) % nheads;
        int s = i / (64 * nheads);

        float inv = __powf(500000.0f, -((float)(2 * d)) * (1.0f / 128.0f));
        float wavelen = 6.283185307179586f / inv;
        float inv_scaled = inv * 0.125f;
        float freq;
        if (wavelen < 2048.0f)      freq = inv;
        else if (wavelen > 8192.0f) freq = inv_scaled;
        else {
            float smooth = (8192.0f / wavelen - 1.0f) * (1.0f / 3.0f);
            freq = (1.0f - smooth) * inv_scaled + smooth * inv;
        }
        float pos = (float)pos_ids[s];
        float ang = pos * freq;
        float c = __cosf(ang), sn = __sinf(ang);

        bf16* p = x + ((size_t)s * nheads + h) * HD;
        float x1 = bf2f(p[d]), x2 = bf2f(p[d + 64]);
        p[d]      = f2bf(x1 * c - x2 * sn);
        p[d + 64] = f2bf(x2 * c + x1 * sn);
    }
}

// ---------------- 4) flash attention (bf16 WMMA, online softmax) ----------
// grid = (S/64, NH); block = 128 (4 waves). Each wave owns a 16-row strip.
// K/V tiles double-buffered; TDM prefetch of next tile overlaps compute.
__global__ __launch_bounds__(128)
void flash_attn(const bf16* __restrict__ Q, const bf16* __restrict__ K,
                const bf16* __restrict__ V, bf16* __restrict__ ctx) {
    __shared__ __align__(16) bf16 Qs[64][HD + 8];
    __shared__ __align__(16) bf16 Ks[2][32][HD + 8];
    __shared__ __align__(16) bf16 Vs[2][32][HD + 8];
    __shared__ __align__(16) bf16 Ps[4][16][40];

    const int tid  = threadIdx.x;
    const int lane = tid & 31;
    const int wave = tid >> 5;            // 0..3
    const int qt   = blockIdx.x;
    const int head = blockIdx.y;
    const int kvh  = head / GQA;
    const int q0   = qt * 64;
    const int jend = q0 + 64;

    // load Q tile (64 x 128) vectorized
    for (int i = tid; i < 64 * (HD / 8); i += 128) {
        int r = i >> 4, c = (i & 15) * 8;
        *(uint4*)&Qs[r][c] = *(const uint4*)&Q[((size_t)(q0 + r) * NH + head) * HD + c];
    }

#if defined(HAVE_TDM)
    if (wave == 0) {   // prologue: stage first K/V tile via TDM
        tdm_load_tile_128x32(&Ks[0][0][0], K + (size_t)kvh * HD, NKV * HD);
        tdm_load_tile_128x32(&Vs[0][0][0], V + (size_t)kvh * HD, NKV * HD);
    }
#endif

    v8f acc[8];
#pragma unroll
    for (int t = 0; t < 8; ++t) acc[t] = (v8f){};
    float mrun[8], lrun[8];
#pragma unroll
    for (int v = 0; v < 8; ++v) { mrun[v] = -1e30f; lrun[v] = 0.0f; }

    const int hi    = lane >> 4;
    const int rbase = q0 + wave * 16 + (hi << 3);

    int it = 0;
    for (int j0 = 0; j0 < jend; j0 += 32, ++it) {
        const int cur = it & 1;
        const bool more = (j0 + 32 < jend);
#if defined(HAVE_TDM)
        if (wave == 0) {
            if (more) {   // prefetch next K/V tile into the other buffer
                tdm_load_tile_128x32(&Ks[cur ^ 1][0][0],
                                     K + ((size_t)(j0 + 32) * NKV + kvh) * HD, NKV * HD);
                tdm_load_tile_128x32(&Vs[cur ^ 1][0][0],
                                     V + ((size_t)(j0 + 32) * NKV + kvh) * HD, NKV * HD);
                __builtin_amdgcn_s_wait_tensorcnt(2);   // newest 2 may stay in flight
            } else {
                __builtin_amdgcn_s_wait_tensorcnt(0);
            }
        }
#else
        for (int i = tid; i < 32 * (HD / 8); i += 128) {
            int r = i >> 4, c = (i & 15) * 8;
            *(uint4*)&Ks[cur][r][c] = *(const uint4*)&K[((size_t)(j0 + r) * NKV + kvh) * HD + c];
            *(uint4*)&Vs[cur][r][c] = *(const uint4*)&V[((size_t)(j0 + r) * NKV + kvh) * HD + c];
        }
#endif
        __syncthreads();

        // S = Q Kt : contraction over HD (4 steps of 32)
        v8f s0 = {}; v8f s1 = {};
#pragma unroll
        for (int kk = 0; kk < 4; ++kk) {
            v16bf aq  = frag_ld(&Qs[wave * 16][kk * 32], HD + 8);
            v16bf bk0 = frag_ld(&Ks[cur][0][kk * 32],  HD + 8);
            v16bf bk1 = frag_ld(&Ks[cur][16][kk * 32], HD + 8);
            s0 = wmma_bf16(aq, bk0, s0);
            s1 = wmma_bf16(aq, bk1, s1);
        }

        // scale + causal mask
        const int col0 = j0 + (lane & 15);
        const int col1 = col0 + 16;
#pragma unroll
        for (int v = 0; v < 8; ++v) {
            int row = rbase + v;
            float a = s0[v] * SCALING, b = s1[v] * SCALING;
            if (col0 > row) a = -1e30f;
            if (col1 > row) b = -1e30f;
            s0[v] = a; s1[v] = b;
        }

        // online softmax (16-lane shuffle reductions stay within halves)
#pragma unroll
        for (int v = 0; v < 8; ++v) {
            float mx = fmaxf(s0[v], s1[v]);
#pragma unroll
            for (int off = 1; off < 16; off <<= 1)
                mx = fmaxf(mx, __shfl_xor(mx, off, 32));
            float mnew = fmaxf(mrun[v], mx);
            float p0 = __expf(s0[v] - mnew);
            float p1 = __expf(s1[v] - mnew);
            float rs = p0 + p1;
#pragma unroll
            for (int off = 1; off < 16; off <<= 1)
                rs += __shfl_xor(rs, off, 32);
            float alpha = __expf(mrun[v] - mnew);
            lrun[v] = lrun[v] * alpha + rs;
            mrun[v] = mnew;
            s0[v] = p0; s1[v] = p1;
#pragma unroll
            for (int t = 0; t < 8; ++t) acc[t][v] *= alpha;
        }

        // P strip (D layout -> LDS row major), reload as A fragment
#pragma unroll
        for (int v = 0; v < 8; ++v) {
            Ps[wave][v + (hi << 3)][lane & 15]        = f2bf(s0[v]);
            Ps[wave][v + (hi << 3)][(lane & 15) + 16] = f2bf(s1[v]);
        }
        v16bf ap = frag_ld(&Ps[wave][0][0], 40);
#pragma unroll
        for (int t = 0; t < 8; ++t) {
            v16bf bv = frag_ld_kn(&Vs[cur][0][t * 16], HD + 8);
            acc[t] = wmma_bf16(ap, bv, acc[t]);
        }
        __syncthreads();   // reads of buffers done before next iter's prefetch lands
    }

    // normalize + store ctx (bf16, [S, NH*HD])
    const int n = lane & 15;
#pragma unroll
    for (int t = 0; t < 8; ++t) {
#pragma unroll
        for (int v = 0; v < 8; ++v) {
            int row = rbase + v;
            float val = acc[t][v] / lrun[v];
            ctx[((size_t)row * NH + head) * HD + t * 16 + n] = f2bf(val);
        }
    }
}

// ---------------- launch ----------------
extern "C" void kernel_launch(void* const* d_in, const int* in_sizes, int n_in,
                              void* d_out, int out_size, void* d_ws, size_t ws_size,
                              hipStream_t stream) {
    const float* hidden = (const float*)d_in[0];
    // d_in[1] = attention_mask (pure causal; applied analytically in-kernel)
    const int*   pos    = (const int*)d_in[2];
    const float* wq     = (const float*)d_in[3];
    const float* wk     = (const float*)d_in[4];
    const float* wv     = (const float*)d_in[5];
    const float* wo     = (const float*)d_in[6];
    float*       out    = (float*)d_out;

    char* ws = (char*)d_ws;
    size_t off = 0;
    auto alloc = [&](size_t bytes) {
        void* p = ws + off;
        off = (off + bytes + 255) & ~(size_t)255;
        return p;
    };
    bf16* hid_bf = (bf16*)alloc((size_t)S_LEN * HID * 2);
    bf16* wq_bf  = (bf16*)alloc((size_t)NH  * HD * HID * 2);
    bf16* wk_bf  = (bf16*)alloc((size_t)NKV * HD * HID * 2);
    bf16* wv_bf  = (bf16*)alloc((size_t)NKV * HD * HID * 2);
    bf16* wo_bf  = (bf16*)alloc((size_t)HID * NH * HD * 2);
    bf16* q_bf   = (bf16*)alloc((size_t)S_LEN * NH  * HD * 2);
    bf16* k_bf   = (bf16*)alloc((size_t)S_LEN * NKV * HD * 2);
    bf16* v_bf   = (bf16*)alloc((size_t)S_LEN * NKV * HD * 2);
    bf16* ctx_bf = (bf16*)alloc((size_t)S_LEN * NH  * HD * 2);

    auto cvt = [&](const float* src, bf16* dst, int n) {
        int n4 = n >> 2;
        int blocks = (n4 + 255) / 256;
        cvt_f32_bf16<<<blocks, 256, 0, stream>>>(src, dst, n4);
    };
    cvt(hidden, hid_bf, S_LEN * HID);
    cvt(wq, wq_bf, NH  * HD * HID);
    cvt(wk, wk_bf, NKV * HD * HID);
    cvt(wv, wv_bf, NKV * HD * HID);
    cvt(wo, wo_bf, HID * NH * HD);

    // Q/K/V projections
    gemm_bf16_wmma<bf16><<<dim3((NH * HD) / 128, S_LEN / 64), 256, 0, stream>>>(
        hid_bf, wq_bf, q_bf, S_LEN, NH * HD, HID);
    gemm_bf16_wmma<bf16><<<dim3((NKV * HD) / 128, S_LEN / 64), 256, 0, stream>>>(
        hid_bf, wk_bf, k_bf, S_LEN, NKV * HD, HID);
    gemm_bf16_wmma<bf16><<<dim3((NKV * HD) / 128, S_LEN / 64), 256, 0, stream>>>(
        hid_bf, wv_bf, v_bf, S_LEN, NKV * HD, HID);

    // RoPE on Q and K
    rope_kernel<<<(S_LEN * NH  * 64 + 255) / 256, 256, 0, stream>>>(q_bf, NH,  pos);
    rope_kernel<<<(S_LEN * NKV * 64 + 255) / 256, 256, 0, stream>>>(k_bf, NKV, pos);

    // flash attention -> ctx
    flash_attn<<<dim3(S_LEN / 64, NH), 128, 0, stream>>>(q_bf, k_bf, v_bf, ctx_bf);

    // output projection -> d_out (fp32)
    gemm_bf16_wmma<float><<<dim3(HID / 128, S_LEN / 64), 256, 0, stream>>>(
        ctx_bf, wo_bf, out, S_LEN, HID, NH * HD);
}
